// PPGN_sub_7842610283375
// MI455X (gfx1250) — matrix operand order repaired and checked by reading, not compiled, not run on hardware
//
#include <hip/hip_runtime.h>
#include <hip/hip_bf16.h>

typedef __attribute__((ext_vector_type(16))) __bf16 v16bf;
typedef __attribute__((ext_vector_type(8)))  __bf16 v8bf;
typedef __attribute__((ext_vector_type(8)))  float  v8f;

#define BSUB  2048
#define SS    16
#define NPOS  256
#define CHN   128
#define EPS   128     // edges per subgraph
#define GNUM  128

union FragU { v16bf v; uint4 q[2]; };

// low 32 bits of a generic pointer to LDS == LDS byte offset
#define LDS_OFF(p) ((unsigned)(unsigned long long)(p))

// ---------------------------------------------------------------------------
// Kernel 0: pre-swizzle an f32 weight matrix [128, ctot] into bf16 WMMA
// A-fragment order: Wp[(kc*8 + m)*32 + lane] = v16bf fragment slice.
// ---------------------------------------------------------------------------
__global__ __launch_bounds__(32)
void pack_w_kernel(const float* __restrict__ W, int ctot, __bf16* __restrict__ Wp)
{
    const int blk  = blockIdx.x;            // kc*8 + m
    const int kc   = blk >> 3, m = blk & 7;
    const int lane = threadIdx.x;
    const int h = lane >> 4, nl = lane & 15;
    const int mrow = m * 16 + nl;
    FragU A;
    #pragma unroll
    for (int v = 0; v < 8; ++v) {
        const int kb = (v < 4) ? (2 * v + 8 * h) : (16 + 2 * (v - 4) + 8 * h);
        const int K0 = kc * 32 + kb;
        A.v[2 * v]     = (__bf16)((K0     < ctot) ? W[mrow * ctot + K0]     : 0.f);
        A.v[2 * v + 1] = (__bf16)((K0 + 1 < ctot) ? W[mrow * ctot + K0 + 1] : 0.f);
    }
    *((v16bf*)Wp + (long)blk * 32 + lane) = A.v;
}

// ---------------------------------------------------------------------------
// Kernel 1: build z0[b, c, q, r] = dea[b, i=r, j=q, c]  (17 channels, bf16)
// ---------------------------------------------------------------------------
__global__ __launch_bounds__(256)
void prep_kernel(const int* __restrict__ edge_index, const float* __restrict__ edge_attr,
                 const float* __restrict__ x, const float* __restrict__ pos,
                 __bf16* __restrict__ z0, int E_total)
{
    __shared__ float ea[5][NPOS];
    __shared__ float ps[SS][3];
    __shared__ float xf[SS][11];
    const int b = blockIdx.x, t = threadIdx.x;
    for (int c = 0; c < 5; ++c) ea[c][t] = 0.f;
    if (t < SS * 3)  ps[t / 3][t % 3]   = pos[(b * SS + t / 3) * 3 + (t % 3)];
    if (t < SS * 11) xf[t / 11][t % 11] = x[(b * SS + t / 11) * 11 + (t % 11)];
    __syncthreads();
    if (t < EPS) {
        const int e  = b * EPS + t;
        const int s  = edge_index[e];
        const int d  = edge_index[E_total + e];
        const int cell = (s & 15) * SS + (d & 15);
        for (int f = 0; f < 4; ++f) atomicAdd(&ea[f][cell], edge_attr[e * 4 + f]);
        atomicAdd(&ea[4][cell], 1.0f);
    }
    __syncthreads();
    const int q = t >> 4, r = t & 15;       // z cell (q, r), p = t
    const int i = r, j = q;                 // dea cell (i, j)
    const long base = (long)b * 17 * NPOS + t;
    for (int c = 0; c < 5; ++c) z0[base + (long)c * NPOS] = (__bf16)ea[c][i * SS + j];
    const float dx = ps[i][0] - ps[j][0];
    const float dy = ps[i][1] - ps[j][1];
    const float dz = ps[i][2] - ps[j][2];
    z0[base + 5L * NPOS] = (__bf16)sqrtf(fmaxf(dx * dx + dy * dy + dz * dz, 0.f));
    for (int c = 0; c < 11; ++c)
        z0[base + (long)(6 + c) * NPOS] = (__bf16)((i == j) ? xf[i][c] : 0.f);
}

// ---------------------------------------------------------------------------
// Kernel 2: batched 1x1-conv GEMM.  OUT[b] = act(W @ concat(IN0,IN1)[b] + bias)
// One workgroup (8 waves) per subgraph; 16 output tiles register-resident.
// Two-deep pipeline using ASYNC global->LDS DMA (no staging VGPRs), coalesced
// epilogue via LDS repack + ASYNC LDS->global stores.  WMMA bf16 -> f32.
// ---------------------------------------------------------------------------
__global__ __launch_bounds__(256)
void conv_gemm(const __bf16* __restrict__ in0, int c0,
               const __bf16* __restrict__ in1, int c1,
               const __bf16* __restrict__ Wp, const float* __restrict__ bias,
               __bf16* __restrict__ out, int do_relu)
{
    __shared__ __bf16 Xs[2][32 * NPOS];        // ping-pong K-chunks, 2 x 16 KB
    __shared__ __bf16 Ws[2][8 * 32 * 16];      // ping-pong A-fragments, 2 x 8 KB
    const int b    = blockIdx.x;
    const int t    = threadIdx.x;
    const int lane = t & 31, wave = t >> 5;
    const int ctot = c0 + c1;
    const int nK   = (ctot + 31) >> 5;
    const int h    = lane >> 4, nl = lane & 15;
    const int krow = t & 31;                   // staged row within chunk
    const int pblk = (t >> 5) * 32;            // 32-position slab

    v8f acc[16];
    #pragma unroll
    for (int i = 0; i < 16; ++i)
        #pragma unroll
        for (int r = 0; r < 8; ++r) acc[i][r] = 0.f;

    auto srcRow = [&](int kk) -> const __bf16* {
        if (kk < c0)   return in0 + ((long)b * c0 + kk) * NPOS;
        if (kk < ctot) return in1 + ((long)b * c1 + (kk - c0)) * NPOS;
        return nullptr;
    };
    // issue async DMA of chunk kc into LDS buffer `buf` (no VGPR data path)
    auto stageAsync = [&](int kc, int buf) {
        const __bf16* src = srcRow(kc * 32 + krow);
        __bf16* xd = &Xs[buf][krow * NPOS + pblk];
        if (src) {
            const unsigned loff = LDS_OFF(xd);
            const __bf16*  g    = src + pblk;
            asm volatile("global_load_async_to_lds_b128 %0, %1, off"
                         :: "v"(loff), "v"(g) : "memory");
            asm volatile("global_load_async_to_lds_b128 %0, %1, off offset:16"
                         :: "v"(loff), "v"(g) : "memory");
            asm volatile("global_load_async_to_lds_b128 %0, %1, off offset:32"
                         :: "v"(loff), "v"(g) : "memory");
            asm volatile("global_load_async_to_lds_b128 %0, %1, off offset:48"
                         :: "v"(loff), "v"(g) : "memory");
        } else {
            uint4* d = (uint4*)xd;
            const uint4 zz = {0u, 0u, 0u, 0u};
            d[0] = zz; d[1] = zz; d[2] = zz; d[3] = zz;
        }
        __bf16* wd = &Ws[buf][(wave * 32 + lane) * 16];
        const unsigned wloff = LDS_OFF(wd);
        const __bf16*  gw    = Wp + (size_t)((kc * 8 + wave) * 32 + lane) * 16;
        asm volatile("global_load_async_to_lds_b128 %0, %1, off"
                     :: "v"(wloff), "v"(gw) : "memory");
        asm volatile("global_load_async_to_lds_b128 %0, %1, off offset:16"
                     :: "v"(wloff), "v"(gw) : "memory");
    };

    stageAsync(0, 0);
    asm volatile("s_wait_asynccnt 0x0" ::: "memory");
    __syncthreads();

    for (int kc = 0; kc < nK; ++kc) {
        const int cur = kc & 1;
        if (kc + 1 < nK) stageAsync(kc + 1, cur ^ 1);   // DMA overlaps WMMAs
        if (kc + 2 < nK) {                              // warm L2 one deeper
            const __bf16* ns = srcRow((kc + 2) * 32 + krow);
            if (ns) __builtin_prefetch(ns + pblk, 0, 1);
        }

        // B fragments for this wave's two N-tiles (N = wave*32 .. +31)
        v16bf B0, B1;
        #pragma unroll
        for (int v = 0; v < 8; ++v) {
            const int k0 = (v < 4) ? (2 * v + 8 * h) : (16 + 2 * (v - 4) + 8 * h);
            B0[2 * v]     = Xs[cur][k0 * NPOS       + wave * 32 + nl];
            B0[2 * v + 1] = Xs[cur][(k0 + 1) * NPOS + wave * 32 + nl];
            B1[2 * v]     = Xs[cur][k0 * NPOS       + wave * 32 + 16 + nl];
            B1[2 * v + 1] = Xs[cur][(k0 + 1) * NPOS + wave * 32 + 16 + nl];
        }
        #pragma unroll
        for (int m = 0; m < 8; ++m) {
            const v16bf A = *(const v16bf*)&Ws[cur][(m * 32 + lane) * 16]; // ds b128 x2
            acc[2 * m]     = __builtin_amdgcn_wmma_f32_16x16x32_bf16(
                                 false, A, false, B0, (short)0, acc[2 * m],     false, false);
            acc[2 * m + 1] = __builtin_amdgcn_wmma_f32_16x16x32_bf16(
                                 false, A, false, B1, (short)0, acc[2 * m + 1], false, false);
        }
        asm volatile("s_wait_asynccnt 0x0" ::: "memory");   // next buffer landed
        __syncthreads();
    }

    // epilogue: bias + (relu), repack through LDS, async LDS->global streams
    __bf16* lpack = &Xs[0][0];                  // 32 KB = 64 ch x 256 pos
    for (int half = 0; half < 2; ++half) {
        #pragma unroll
        for (int mm = 0; mm < 4; ++mm) {
            const int m = half * 4 + mm;
            for (int tt = 0; tt < 2; ++tt)
                #pragma unroll
                for (int r = 0; r < 8; ++r) {
                    const int chl = mm * 16 + r + 8 * h;
                    const int p   = wave * 32 + tt * 16 + nl;
                    float vv = acc[2 * m + tt][r] + bias[half * 64 + chl];
                    if (do_relu) vv = fmaxf(vv, 0.f);
                    lpack[chl * NPOS + p] = (__bf16)vv;
                }
        }
        __syncthreads();
        {
            const unsigned soff = LDS_OFF(lpack + t * 64);
            __bf16* g = out + ((long)b * CHN + half * 64) * NPOS + t * 64;
            #define ASTORE(OFF) \
                asm volatile("global_store_async_from_lds_b128 %0, %1, off offset:" #OFF \
                             :: "v"(g), "v"(soff) : "memory")
            ASTORE(0);  ASTORE(16); ASTORE(32); ASTORE(48);
            ASTORE(64); ASTORE(80); ASTORE(96); ASTORE(112);
            #undef ASTORE
            asm volatile("s_wait_asynccnt 0x0" ::: "memory");
        }
        __syncthreads();
    }
}

// ---------------------------------------------------------------------------
// Kernel 3: per-channel 16x16x16 matmul, one WMMA per channel (K padded to 32)
// ---------------------------------------------------------------------------
__global__ __launch_bounds__(256)
void mult_kernel(const __bf16* __restrict__ Am, const __bf16* __restrict__ Bm,
                 __bf16* __restrict__ out)
{
    const int b = blockIdx.x, t = threadIdx.x;
    const int lane = t & 31, wave = t >> 5;
    const int h = lane >> 4, nl = lane & 15;
    for (int cc = 0; cc < 16; ++cc) {
        const int ch = wave * 16 + cc;
        const long base = ((long)b * CHN + ch) * NPOS;
        // A: M = nl (row i), K = 8h + e for e<8 -> contiguous 16B load
        FragU Af;
        Af.q[0] = *(const uint4*)(Am + base + nl * 16 + 8 * h);
        Af.q[1] = make_uint4(0u, 0u, 0u, 0u);
        // B: K = row k (< 16), N = nl (col j); upper K half zero
        v16bf Bf;
        #pragma unroll
        for (int v = 0; v < 8; ++v) {
            const int k0 = (v < 4) ? (2 * v + 8 * h) : (16 + 2 * (v - 4) + 8 * h);
            if (k0 < 16) {
                Bf[2 * v]     = Bm[base + k0 * 16 + nl];
                Bf[2 * v + 1] = Bm[base + (k0 + 1) * 16 + nl];
            } else { Bf[2 * v] = (__bf16)0.f; Bf[2 * v + 1] = (__bf16)0.f; }
        }
        v8f acc;
        #pragma unroll
        for (int r = 0; r < 8; ++r) acc[r] = 0.f;
        acc = __builtin_amdgcn_wmma_f32_16x16x32_bf16(false, Af.v, false, Bf, (short)0,
                                                      acc, false, false);
        #pragma unroll
        for (int r = 0; r < 8; ++r)
            out[base + (r + 8 * h) * 16 + nl] = (__bf16)acc[r];
    }
}

// ---------------------------------------------------------------------------
// Kernel 4: per-(b,c) diag max / offdiag max / global-min partials
// ---------------------------------------------------------------------------
__global__ __launch_bounds__(128)
void stat_kernel(const __bf16* __restrict__ z, float* __restrict__ maxdiag,
                 float* __restrict__ maxoff, float* __restrict__ sdiag,
                 float* __restrict__ smin)
{
    __shared__ float red[128], red2[128];
    const int b = blockIdx.x, c = threadIdx.x;
    const long base = ((long)b * CHN + c) * NPOS;
    float dmax = -1e30f, omax = -1e30f, amin = 1e30f;
    for (int p8 = 0; p8 < NPOS / 8; ++p8) {
        const v8bf vv = *(const v8bf*)(z + base + p8 * 8);
        #pragma unroll
        for (int u = 0; u < 8; ++u) {
            const int p = p8 * 8 + u;
            const float v = (float)vv[u];
            if ((p >> 4) == (p & 15)) dmax = fmaxf(dmax, v);
            else                      omax = fmaxf(omax, v);
            amin = fminf(amin, v);
        }
    }
    maxdiag[b * CHN + c] = dmax;
    maxoff[b * CHN + c]  = omax;
    red[c] = dmax; red2[c] = amin;
    __syncthreads();
    for (int s = 64; s > 0; s >>= 1) {
        if (c < s) { red[c] = fmaxf(red[c], red[c + s]); red2[c] = fminf(red2[c], red2[c + s]); }
        __syncthreads();
    }
    if (c == 0) { sdiag[b] = red[0]; smin[b] = red2[0]; }
}

__global__ __launch_bounds__(256)
void val_kernel(const float* __restrict__ sdiag, const float* __restrict__ smin,
                float* __restrict__ valout)
{
    __shared__ float r1[256], r2[256];
    const int t = threadIdx.x;
    float a = -1e30f, mn = 1e30f;
    for (int b = t; b < BSUB; b += 256) { a = fmaxf(a, sdiag[b]); mn = fminf(mn, smin[b]); }
    r1[t] = a; r2[t] = mn;
    __syncthreads();
    for (int s = 128; s > 0; s >>= 1) {
        if (t < s) { r1[t] = fmaxf(r1[t], r1[t + s]); r2[t] = fminf(r2[t], r2[t + s]); }
        __syncthreads();
    }
    if (t == 0) valout[0] = fabsf(r1[0] - r2[0]);   // |max_diag.max() + (-z).max()|
}

// ---------------------------------------------------------------------------
// Kernel 5: h[b] = relu(fc_w @ [maxdiag ; max(maxoff, maxdiag - val)] + fc_b)
// ---------------------------------------------------------------------------
__global__ __launch_bounds__(64)
void fc_kernel(const float* __restrict__ maxdiag, const float* __restrict__ maxoff,
               const float* __restrict__ valp, const float* __restrict__ fcw,
               const float* __restrict__ fcb, float* __restrict__ hout)
{
    __shared__ float feat[2 * CHN];
    const int b = blockIdx.x, t = threadIdx.x;
    const float val = valp[0];
    for (int c = t; c < CHN; c += 64) {
        const float md = maxdiag[b * CHN + c];
        feat[c]       = md;
        feat[CHN + c] = fmaxf(maxoff[b * CHN + c], md - val);
    }
    __syncthreads();
    float s = fcb[t];
    for (int k = 0; k < 2 * CHN; ++k) s += fcw[t * 2 * CHN + k] * feat[k];
    hout[b * 64 + t] = fmaxf(s, 0.f);
}

// ---------------------------------------------------------------------------
// Kernel 6: per-graph mean (16 subgraphs) + elu(fc1) + elu(fc2) + fc3
// ---------------------------------------------------------------------------
__global__ __launch_bounds__(64)
void head_kernel(const float* __restrict__ h,
                 const float* __restrict__ fc1w, const float* __restrict__ fc1b,
                 const float* __restrict__ fc2w, const float* __restrict__ fc2b,
                 const float* __restrict__ fc3w, const float* __restrict__ fc3b,
                 float* __restrict__ outp)
{
    __shared__ float hm[64], y1[32], y2[16];
    const int g = blockIdx.x, t = threadIdx.x;
    float s = 0.f;
    for (int k = 0; k < 16; ++k) s += h[((long)g * 16 + k) * 64 + t];
    hm[t] = s * (1.f / 16.f);
    __syncthreads();
    if (t < 32) {
        float a = fc1b[t];
        for (int k = 0; k < 64; ++k) a += fc1w[t * 64 + k] * hm[k];
        y1[t] = (a > 0.f) ? a : (expf(a) - 1.f);
    }
    __syncthreads();
    if (t < 16) {
        float a = fc2b[t];
        for (int k = 0; k < 32; ++k) a += fc2w[t * 32 + k] * y1[k];
        y2[t] = (a > 0.f) ? a : (expf(a) - 1.f);
    }
    __syncthreads();
    if (t == 0) {
        float a = fc3b[0];
        for (int k = 0; k < 16; ++k) a += fc3w[k] * y2[k];
        outp[g] = a;
    }
}

// ---------------------------------------------------------------------------
extern "C" void kernel_launch(void* const* d_in, const int* in_sizes, int n_in,
                              void* d_out, int out_size, void* d_ws, size_t ws_size,
                              hipStream_t stream)
{
    const int*   edge_index = (const int*)d_in[0];
    const float* edge_attr  = (const float*)d_in[1];
    const float* x          = (const float*)d_in[2];
    const float* pos        = (const float*)d_in[3];
    const float* rb1_m1_w0 = (const float*)d_in[6];
    const float* rb1_m1_b0 = (const float*)d_in[7];
    const float* rb1_m1_w1 = (const float*)d_in[8];
    const float* rb1_m1_b1 = (const float*)d_in[9];
    const float* rb1_m2_w0 = (const float*)d_in[10];
    const float* rb1_m2_b0 = (const float*)d_in[11];
    const float* rb1_m2_w1 = (const float*)d_in[12];
    const float* rb1_m2_b1 = (const float*)d_in[13];
    const float* rb1_sk_w  = (const float*)d_in[14];
    const float* rb1_sk_b  = (const float*)d_in[15];
    const float* rb2_m1_w0 = (const float*)d_in[16];
    const float* rb2_m1_b0 = (const float*)d_in[17];
    const float* rb2_m1_w1 = (const float*)d_in[18];
    const float* rb2_m1_b1 = (const float*)d_in[19];
    const float* rb2_m2_w0 = (const float*)d_in[20];
    const float* rb2_m2_b0 = (const float*)d_in[21];
    const float* rb2_m2_w1 = (const float*)d_in[22];
    const float* rb2_m2_b1 = (const float*)d_in[23];
    const float* rb2_sk_w  = (const float*)d_in[24];
    const float* rb2_sk_b  = (const float*)d_in[25];
    const float* fc_w  = (const float*)d_in[26];
    const float* fc_b  = (const float*)d_in[27];
    const float* fc1_w = (const float*)d_in[28];
    const float* fc1_b = (const float*)d_in[29];
    const float* fc2_w = (const float*)d_in[30];
    const float* fc2_b = (const float*)d_in[31];
    const float* fc3_w = (const float*)d_in[32];
    const float* fc3_b = (const float*)d_in[33];

    const int E = in_sizes[1] / 4;

    char* ws = (char*)d_ws;
    size_t off = 0;
    __bf16* Z0 = (__bf16*)(ws + off);
    off += (size_t)BSUB * 17 * NPOS * sizeof(__bf16);
    off = (off + 255) & ~(size_t)255;
    const size_t big = (size_t)BSUB * CHN * NPOS * sizeof(__bf16);   // 128 MiB
    __bf16* R1 = (__bf16*)(ws + off); off += big;
    __bf16* R2 = (__bf16*)(ws + off); off += big;
    __bf16* R3 = (__bf16*)(ws + off); off += big;
    __bf16* R4 = (__bf16*)(ws + off); off += big;
    float* maxdiag = (float*)(ws + off); off += (size_t)BSUB * CHN * 4;
    float* maxoff  = (float*)(ws + off); off += (size_t)BSUB * CHN * 4;
    float* sdiag   = (float*)(ws + off); off += (size_t)BSUB * 4;
    float* sminb   = (float*)(ws + off); off += (size_t)BSUB * 4;
    float* valp    = (float*)(ws + off); off += 256;
    float* hbuf    = (float*)(ws + off); off += (size_t)BSUB * 64 * 4;

    // packed bf16 weight fragments (nK * 8 KiB per matrix)
    auto alloc_pack = [&](int ctot) {
        const int nK = (ctot + 31) >> 5;
        off = (off + 255) & ~(size_t)255;
        __bf16* p = (__bf16*)(ws + off);
        off += (size_t)nK * 8 * 32 * 16 * sizeof(__bf16);
        return p;
    };
    __bf16* P_r1m1w0 = alloc_pack(17);
    __bf16* P_r1m1w1 = alloc_pack(128);
    __bf16* P_r1m2w0 = alloc_pack(17);
    __bf16* P_r1m2w1 = alloc_pack(128);
    __bf16* P_r1sk   = alloc_pack(145);
    __bf16* P_r2m1w0 = alloc_pack(128);
    __bf16* P_r2m1w1 = alloc_pack(128);
    __bf16* P_r2m2w0 = alloc_pack(128);
    __bf16* P_r2m2w1 = alloc_pack(128);
    __bf16* P_r2sk   = alloc_pack(256);

    auto pack = [&](const float* W, int ctot, __bf16* P) {
        const int nK = (ctot + 31) >> 5;
        pack_w_kernel<<<nK * 8, 32, 0, stream>>>(W, ctot, P);
    };
    pack(rb1_m1_w0, 17, P_r1m1w0);   pack(rb1_m1_w1, 128, P_r1m1w1);
    pack(rb1_m2_w0, 17, P_r1m2w0);   pack(rb1_m2_w1, 128, P_r1m2w1);
    pack(rb1_sk_w, 145, P_r1sk);
    pack(rb2_m1_w0, 128, P_r2m1w0);  pack(rb2_m1_w1, 128, P_r2m1w1);
    pack(rb2_m2_w0, 128, P_r2m2w0);  pack(rb2_m2_w1, 128, P_r2m2w1);
    pack(rb2_sk_w, 256, P_r2sk);

    prep_kernel<<<BSUB, 256, 0, stream>>>(edge_index, edge_attr, x, pos, Z0, E);

    // ---- regular block 1 (z = Z0, Cin = 17) ----
    conv_gemm<<<BSUB, 256, 0, stream>>>(Z0, 17, nullptr, 0, P_r1m1w0, rb1_m1_b0, R1, 1);
    conv_gemm<<<BSUB, 256, 0, stream>>>(R1, 128, nullptr, 0, P_r1m1w1, rb1_m1_b1, R2, 1); // a
    conv_gemm<<<BSUB, 256, 0, stream>>>(Z0, 17, nullptr, 0, P_r1m2w0, rb1_m2_b0, R1, 1);
    conv_gemm<<<BSUB, 256, 0, stream>>>(R1, 128, nullptr, 0, P_r1m2w1, rb1_m2_b1, R3, 1); // b
    mult_kernel<<<BSUB, 256, 0, stream>>>(R2, R3, R1);                                    // mult
    conv_gemm<<<BSUB, 256, 0, stream>>>(Z0, 17, R1, 128, P_r1sk, rb1_sk_b, R4, 0);        // z1

    // ---- regular block 2 (z = R4, Cin = 128) ----
    conv_gemm<<<BSUB, 256, 0, stream>>>(R4, 128, nullptr, 0, P_r2m1w0, rb2_m1_b0, R2, 1);
    conv_gemm<<<BSUB, 256, 0, stream>>>(R2, 128, nullptr, 0, P_r2m1w1, rb2_m1_b1, R3, 1); // a
    conv_gemm<<<BSUB, 256, 0, stream>>>(R4, 128, nullptr, 0, P_r2m2w0, rb2_m2_b0, R2, 1);
    conv_gemm<<<BSUB, 256, 0, stream>>>(R2, 128, nullptr, 0, P_r2m2w1, rb2_m2_b1, R1, 1); // b
    mult_kernel<<<BSUB, 256, 0, stream>>>(R3, R1, R2);                                    // mult
    conv_gemm<<<BSUB, 256, 0, stream>>>(R4, 128, R2, 128, P_r2sk, rb2_sk_b, R1, 0);       // z2

    // ---- pooling + head ----
    stat_kernel<<<BSUB, 128, 0, stream>>>(R1, maxdiag, maxoff, sdiag, sminb);
    val_kernel<<<1, 256, 0, stream>>>(sdiag, sminb, valp);
    fc_kernel<<<BSUB, 64, 0, stream>>>(maxdiag, maxoff, valp, fc_w, fc_b, hbuf);
    head_kernel<<<GNUM, 64, 0, stream>>>(hbuf, fc1_w, fc1_b, fc2_w, fc2_b, fc3_w, fc3_b,
                                         (float*)d_out);
}
